// PositionwiseFeedForward_19267223290743
// MI455X (gfx1250) — 8-bit matrix operand order repaired and checked by reading, compiled, not run on hardware
//
#include <hip/hip_runtime.h>
#include <cstddef>
#include <cstdint>

typedef __attribute__((ext_vector_type(8))) int          v8i;
typedef __attribute__((ext_vector_type(4))) unsigned int v4u;
typedef __attribute__((ext_vector_type(2))) unsigned int v2u;

#define EPSQ 1e-8f

#if __has_builtin(__builtin_amdgcn_global_load_async_to_lds_b128) && \
    __has_builtin(__builtin_amdgcn_s_wait_asynccnt)
#define HAVE_ASYNC 1
typedef int v4i_n __attribute__((vector_size(16)));
typedef __attribute__((address_space(1))) v4i_n v4i_g;
typedef __attribute__((address_space(3))) v4i_n v4i_l;
#define TO_G(p) ((v4i_g*)(unsigned long long)(const void*)(p))
#define TO_L(p) ((v4i_l*)(unsigned int)(unsigned long long)(void*)(p))
#else
#define HAVE_ASYNC 0
#endif

__global__ void reduce_max_stage1(const float* __restrict__ x, size_t n,
                                  float* __restrict__ part) {
  __shared__ float sm[256];
  float m = -3.4e38f;
  size_t stride = (size_t)gridDim.x * blockDim.x;
  for (size_t i = (size_t)blockIdx.x * blockDim.x + threadIdx.x; i < n; i += stride)
    m = fmaxf(m, x[i]);
  sm[threadIdx.x] = m;
  __syncthreads();
  for (int s = 128; s > 0; s >>= 1) {
    if ((int)threadIdx.x < s) sm[threadIdx.x] = fmaxf(sm[threadIdx.x], sm[threadIdx.x + s]);
    __syncthreads();
  }
  if (threadIdx.x == 0) part[blockIdx.x] = sm[0];
}

__global__ void reduce_max_stage2(const float* __restrict__ part, int n,
                                  float* __restrict__ sOut, float denom) {
  __shared__ float sm[256];
  float m = -3.4e38f;
  for (int i = threadIdx.x; i < n; i += 256) m = fmaxf(m, part[i]);
  sm[threadIdx.x] = m;
  __syncthreads();
  for (int s = 128; s > 0; s >>= 1) {
    if ((int)threadIdx.x < s) sm[threadIdx.x] = fmaxf(sm[threadIdx.x], sm[threadIdx.x + s]);
    __syncthreads();
  }
  if (threadIdx.x == 0) sOut[0] = fmaxf(sm[0], EPSQ) / denom;
}

__global__ void quant_weight_kernel(const float* __restrict__ w,
                                    signed char* __restrict__ wq,
                                    float* __restrict__ sW, int C, int K) {
  int wave = threadIdx.x >> 5;
  int lane = threadIdx.x & 31;
  int ch = blockIdx.x * 8 + wave;
  if (ch >= C) return;
  const float* row = w + (size_t)ch * K;
  float m = 0.0f;
  for (int i = lane; i < K; i += 32) m = fmaxf(m, fabsf(row[i]));
  for (int off = 16; off >= 1; off >>= 1) m = fmaxf(m, __shfl_xor(m, off, 32));
  float s = fmaxf(m, EPSQ) / 127.0f;
  if (lane == 0) sW[ch] = s;
  float inv = 1.0f / s;
  signed char* qrow = wq + (size_t)ch * K;
  for (int i = lane; i < K; i += 32) {
    float q = fminf(fmaxf(rintf(row[i] * inv), -128.0f), 127.0f);
    qrow[i] = (signed char)q;
  }
}

__global__ void quant_act_kernel(const float* __restrict__ x,
                                 unsigned char* __restrict__ xq,
                                 const float* __restrict__ sPtr, size_t n) {
  float inv = 1.0f / sPtr[0];
  size_t stride = (size_t)gridDim.x * blockDim.x;
  for (size_t i = (size_t)blockIdx.x * blockDim.x + threadIdx.x; i < n; i += stride) {
    float q = fminf(fmaxf(rintf(x[i] * inv), 0.0f), 255.0f);
    xq[i] = (unsigned char)q;
  }
}

#define LDS_STRIDE 80

template <bool RELU>
__global__ __launch_bounds__(256)
void gemm_iu8_kernel(const unsigned char* __restrict__ A,
                     const signed char* __restrict__ W,
                     const float* __restrict__ sInPtr,
                     const float* __restrict__ sW,
                     const float* __restrict__ bias,
                     float* __restrict__ out,
                     float* __restrict__ partMax,
                     int M, int N, int K) {
  __shared__ unsigned char lA[2][128 * LDS_STRIDE];
  __shared__ unsigned char lB[2][128 * LDS_STRIDE];

  const int t    = threadIdx.x;
  const int wave = t >> 5;
  const int lane = t & 31;
  const int wm   = wave >> 2;
  const int wn   = wave & 3;
  const int half = lane >> 4;
  const int l16  = lane & 15;

  const int blockM = blockIdx.y * 128;
  const int blockN = blockIdx.x * 128;

  const int stgRow = t >> 1;
  const int stgSeg = (t & 1) * 32;
  const int stgOff = stgRow * LDS_STRIDE + stgSeg;
  const unsigned char* gA = A + (size_t)(blockM + stgRow) * K + stgSeg;
  const unsigned char* gB = (const unsigned char*)W + (size_t)(blockN + stgRow) * K + stgSeg;

  v8i acc[4][2] = {};
  int cur = 0;

#if HAVE_ASYNC
  auto stage = [&](int k0, int buf) {
    __builtin_amdgcn_global_load_async_to_lds_b128(TO_G(gA + k0),      TO_L(&lA[buf][stgOff]),      0, 0);
    __builtin_amdgcn_global_load_async_to_lds_b128(TO_G(gA + k0 + 16), TO_L(&lA[buf][stgOff + 16]), 0, 0);
    __builtin_amdgcn_global_load_async_to_lds_b128(TO_G(gB + k0),      TO_L(&lB[buf][stgOff]),      0, 0);
    __builtin_amdgcn_global_load_async_to_lds_b128(TO_G(gB + k0 + 16), TO_L(&lB[buf][stgOff + 16]), 0, 0);
  };
  stage(0, 0);
#else
  v4u rA0, rA1, rB0, rB1;
  auto loadTile = [&](int k0) {
    rA0 = *(const v4u*)(gA + k0);
    rA1 = *(const v4u*)(gA + k0 + 16);
    rB0 = *(const v4u*)(gB + k0);
    rB1 = *(const v4u*)(gB + k0 + 16);
    if (k0 + 64 < K) {
      __builtin_prefetch(gA + k0 + 64, 0, 3);
      __builtin_prefetch(gB + k0 + 64, 0, 3);
    }
  };
  auto storeTile = [&](int buf) {
    *(v4u*)(&lA[buf][stgOff])      = rA0;
    *(v4u*)(&lA[buf][stgOff + 16]) = rA1;
    *(v4u*)(&lB[buf][stgOff])      = rB0;
    *(v4u*)(&lB[buf][stgOff + 16]) = rB1;
  };
  loadTile(0);
  storeTile(0);
#endif

  for (int k0 = 0; k0 < K; k0 += 64) {
    const bool hasNext = (k0 + 64) < K;
#if HAVE_ASYNC
    __builtin_amdgcn_s_wait_asynccnt(0);
    __syncthreads();
    if (hasNext) stage(k0 + 64, cur ^ 1);
#else
    __syncthreads();
    if (hasNext) loadTile(k0 + 64);
#endif

    v8i afrag[4];
    for (int mi = 0; mi < 4; ++mi) {
      const unsigned char* ar = &lA[cur][(wm * 64 + mi * 16 + l16) * LDS_STRIDE + half * 16];
      v4u t0 = *(const v4u*)(ar +  0);
      v4u t1 = *(const v4u*)(ar + 32);
      v8i a;
      a[0] = (int)t0.x; a[1] = (int)t0.y;
      a[2] = (int)t0.z; a[3] = (int)t0.w;
      a[4] = (int)t1.x; a[5] = (int)t1.y;
      a[6] = (int)t1.z; a[7] = (int)t1.w;
      afrag[mi] = a;
    }

    v8i bfrag[2];
    for (int ni = 0; ni < 2; ++ni) {
      const unsigned char* br = &lB[cur][(wn * 32 + ni * 16 + l16) * LDS_STRIDE + half * 16];
      v4u u0 = *(const v4u*)(br +  0);
      v4u u1 = *(const v4u*)(br + 32);
      v8i b;
      b[0] = (int)u0.x; b[1] = (int)u0.y; b[2] = (int)u0.z; b[3] = (int)u0.w;
      b[4] = (int)u1.x; b[5] = (int)u1.y; b[6] = (int)u1.z; b[7] = (int)u1.w;
      bfrag[ni] = b;
    }

    for (int mi = 0; mi < 4; ++mi)
      for (int ni = 0; ni < 2; ++ni)
        acc[mi][ni] = __builtin_amdgcn_wmma_i32_16x16x64_iu8(
             false, afrag[mi],
             true,  bfrag[ni],
            acc[mi][ni],  false,  false);

#if !HAVE_ASYNC
    if (hasNext) storeTile(cur ^ 1);
#endif
    cur ^= 1;
  }

  const float sIn = sInPtr[0];
  float localMax = 0.0f;
  for (int ni = 0; ni < 2; ++ni) {
    int n = blockN + wn * 32 + ni * 16 + l16;
    float sp = sIn * sW[n];
    float bq = rintf(bias[n] / sp) * sp;
    for (int mi = 0; mi < 4; ++mi) {
      int mBase = blockM + wm * 64 + mi * 16 + half * 8;
      for (int v = 0; v < 8; ++v) {
        float val = (float)acc[mi][ni][v] * sp + bq;
        if (RELU) {
          val = fmaxf(val, 0.0f);
          localMax = fmaxf(localMax, val);
        }
        out[(size_t)(mBase + v) * N + n] = val;
      }
    }
  }

  if (RELU) {
    __syncthreads();
    float* red = (float*)&lA[0][0];
    red[t] = localMax;
    __syncthreads();
    for (int s = 128; s > 0; s >>= 1) {
      if (t < s) red[t] = fmaxf(red[t], red[t + s]);
      __syncthreads();
    }
    if (t == 0) partMax[blockIdx.y * gridDim.x + blockIdx.x] = red[0];
  }
}

extern "C" void kernel_launch(void* const* d_in, const int* in_sizes, int n_in,
                              void* d_out, int out_size, void* d_ws, size_t ws_size,
                              hipStream_t stream) {
  (void)in_sizes; (void)n_in; (void)out_size; (void)ws_size;

  const float* x  = (const float*)d_in[0];
  const float* w1 = (const float*)d_in[1];
  const float* b1 = (const float*)d_in[2];
  const float* w2 = (const float*)d_in[3];
  const float* b2 = (const float*)d_in[4];
  float* out = (float*)d_out;

  const int M = 4 * 2048;
  const int D = 1024;
  const int F = 4096;
  const int G1_BLOCKS = (F / 128) * (M / 128);

  size_t off = 0;
  auto alloc = [&](size_t bytes) -> void* {
    off = (off + 255) & ~(size_t)255;
    void* p = (char*)d_ws + off;
    off += bytes;
    return p;
  };
  unsigned char* xq  = (unsigned char*)alloc((size_t)M * D);
  signed char*   wq1 = (signed char*)  alloc((size_t)F * D);
  signed char*   wq2 = (signed char*)  alloc((size_t)D * F);
  float*         h   = (float*)        alloc((size_t)M * F * sizeof(float));
  unsigned char* hq  = (unsigned char*)alloc((size_t)M * F);
  float* partials = (float*)alloc((size_t)G1_BLOCKS * sizeof(float));
  float* sIn1     = (float*)alloc(sizeof(float));
  float* sW1      = (float*)alloc((size_t)F * sizeof(float));
  float* sIn2     = (float*)alloc(sizeof(float));
  float* sW2      = (float*)alloc((size_t)D * sizeof(float));

  reduce_max_stage1<<<1024, 256, 0, stream>>>(x, (size_t)M * D, partials);
  reduce_max_stage2<<<1, 256, 0, stream>>>(partials, 1024, sIn1, 255.0f);
  quant_act_kernel<<<1024, 256, 0, stream>>>(x, xq, sIn1, (size_t)M * D);
  quant_weight_kernel<<<F / 8, 256, 0, stream>>>(w1, wq1, sW1, F, D);
  quant_weight_kernel<<<D / 8, 256, 0, stream>>>(w2, wq2, sW2, D, F);

  gemm_iu8_kernel<true><<<dim3(F / 128, M / 128), 256, 0, stream>>>(
      xq, wq1, sIn1, sW1, b1, h, partials, M, F, D);

  reduce_max_stage2<<<1, 256, 0, stream>>>(partials, G1_BLOCKS, sIn2, 255.0f);
  quant_act_kernel<<<1024, 256, 0, stream>>>(h, hq, sIn2, (size_t)M * F);

  gemm_iu8_kernel<false><<<dim3(D / 128, M / 128), 256, 0, stream>>>(
      hq, wq2, sIn2, sW2, b2, out, nullptr, M, D, F);
}
